// AwA2Conv_75488345194646
// MI455X (gfx1250) — matrix operand order repaired and checked by reading, not compiled, not run on hardware
//
#include <hip/hip_runtime.h>
#include <stdint.h>

typedef __attribute__((ext_vector_type(16))) _Float16 v16h;
typedef __attribute__((ext_vector_type(8)))  _Float16 v8h;
typedef __attribute__((ext_vector_type(8)))  float    v8f;

#define N_NODES 50000
#define N_EDGES 250000
#define N_PAD   50048   /* 128-row padded node count (391 * 128) */

// ---------------------------------------------------------------- utilities
__global__ void fill_f32(float* __restrict__ p, float v, long n) {
  long i = (long)blockIdx.x * blockDim.x + threadIdx.x;
  if (i < n) p[i] = v;
}

__global__ void degree_kernel(const int* __restrict__ src, const int* __restrict__ dst,
                              float* __restrict__ degO, float* __restrict__ degI) {
  int e = blockIdx.x * blockDim.x + threadIdx.x;
  if (e < N_EDGES) {
    atomicAdd(&degO[src[e]], 1.0f);
    atomicAdd(&degI[dst[e]], 1.0f);
  }
}

__global__ void rsqrt_kernel(float* __restrict__ d, int n) {
  int i = blockIdx.x * blockDim.x + threadIdx.x;
  if (i < n) d[i] = rsqrtf(fmaxf(d[i], 1.0f));
}

// layer-1 staging: A[row][k] = (row<N && k<fi) ? x[row,k]*dinvO[row] : 0   (f16)
__global__ void stage0_kernel(const float* __restrict__ x, const float* __restrict__ dinvO,
                              _Float16* __restrict__ A, int fi, int fiPad) {
  long i = (long)blockIdx.x * blockDim.x + threadIdx.x;
  long total = (long)N_PAD * fiPad;
  if (i >= total) return;
  int row = (int)(i / fiPad);
  int k   = (int)(i % fiPad);
  float v = 0.0f;
  if (row < N_NODES && k < fi) v = x[(long)row * fi + k] * dinvO[row];
  A[i] = (_Float16)v;
}

// weight convert: Wt[n][k] = (k<fi) ? W[k,n] : 0   (f16, transposed, K padded)
__global__ void wt_kernel(const float* __restrict__ W, _Float16* __restrict__ Wt,
                          int fi, int fo, int fiPad) {
  long i = (long)blockIdx.x * blockDim.x + threadIdx.x;
  long total = (long)fo * fiPad;
  if (i >= total) return;
  int n = (int)(i / fiPad);
  int k = (int)(i % fiPad);
  Wt[i] = (_Float16)((k < fi) ? W[(long)k * fo + n] : 0.0f);
}

// ---------------------------------------------------------------- async copy
// CDNA5 GLOBAL_LOAD_ASYNC_TO_LDS_B128: per-lane 16B global -> LDS, no VGPR
// data path, tracked with ASYNCcnt.  lds_off = wave-relative LDS byte address
// (generic LDS pointers carry it in bits [31:0]).
__device__ __forceinline__ void async_ld16(uint32_t lds_off, const _Float16* gptr) {
  asm volatile("global_load_async_to_lds_b128 %0, %1, off"
               : : "v"(lds_off), "v"(gptr) : "memory");
}
__device__ __forceinline__ uint32_t lds_off_of(const void* p) {
  return (uint32_t)(uintptr_t)p;
}

// ---------------------------------------------------------------- WMMA GEMM
// C[M_PAD, N] = A[M_PAD, K] * Bt[N, K]^T   (f16 inputs, f32 accumulate)
// block = 256 threads (8 waves); tile = 128 rows x 64 cols; K step = 32.
// Double-buffered LDS, async global->LDS pipelined behind WMMA.
#define SA_STRIDE 40
#define SA_ELEMS  (128 * SA_STRIDE)
#define SB_ELEMS  (64 * SA_STRIDE)

__global__ __launch_bounds__(256)
void gemm_wmma_kernel(const _Float16* __restrict__ A, const _Float16* __restrict__ Bt,
                      float* __restrict__ C, int K, int N) {
  __shared__ _Float16 sA[2 * SA_ELEMS];
  __shared__ _Float16 sB[2 * SB_ELEMS];

  const int tid  = threadIdx.x;
  const int wave = tid >> 5;
  const int lane = tid & 31;
  const int l16  = lane & 15;
  const int kh   = lane >> 4;          // which K-half this lane holds
  const long rowBase = (long)blockIdx.x * 128;
  const int  colBase = blockIdx.y * 64;

  // per-thread cooperative-load coordinates (3 async 16B chunks per K-tile)
  const int aRow0 = tid >> 2,          aCh0 = tid & 3;          // A chunks 0..255
  const int aRow1 = (tid + 256) >> 2,  aCh1 = tid & 3;          // A chunks 256..511
  const int bRow  = tid >> 2,          bCh  = tid & 3;          // B chunks 0..255

  const _Float16* gA0 = A + (rowBase + aRow0) * (long)K + aCh0 * 8;
  const _Float16* gA1 = A + (rowBase + aRow1) * (long)K + aCh1 * 8;
  const _Float16* gB  = Bt + (long)(colBase + bRow) * K + bCh * 8;

  const uint32_t lA0 = lds_off_of(sA + aRow0 * SA_STRIDE + aCh0 * 8);
  const uint32_t lA1 = lds_off_of(sA + aRow1 * SA_STRIDE + aCh1 * 8);
  const uint32_t lB  = lds_off_of(sB + bRow * SA_STRIDE + bCh * 8);
  const uint32_t bufStepA = SA_ELEMS * 2;   // bytes between the two A buffers
  const uint32_t bufStepB = SB_ELEMS * 2;   // bytes between the two B buffers

  v8f acc[4] = {v8f{}, v8f{}, v8f{}, v8f{}};

  // prologue: async-fetch K-tile 0 into buffer 0
  async_ld16(lA0, gA0);
  async_ld16(lA1, gA1);
  async_ld16(lB,  gB);

  int buf = 0;
  for (int kb = 0; kb < K; kb += 32, buf ^= 1) {
    if (kb + 32 < K) {
      // fetch next K-tile into the other buffer (protected by previous
      // iteration's trailing barrier), then wait for the current tile only.
      const uint32_t bo = (uint32_t)(buf ^ 1);
      async_ld16(lA0 + bo * bufStepA, gA0 + kb + 32);
      async_ld16(lA1 + bo * bufStepA, gA1 + kb + 32);
      async_ld16(lB  + bo * bufStepB, gB  + kb + 32);
      asm volatile("s_wait_asynccnt 0x3" ::: "memory");
    } else {
      asm volatile("s_wait_asynccnt 0x0" ::: "memory");
    }
    __syncthreads();

    const _Float16* sAb = sA + buf * SA_ELEMS;
    const _Float16* sBb = sB + buf * SB_ELEMS;

    // A fragment (ISA 16-bit A layout): lane(l16,kh) holds row m,
    // elements 0..7 = K kh*8..kh*8+7, elements 8..15 = K 16+kh*8..
    int m = wave * 16 + l16;
    v8h alo = *(const v8h*)(sAb + m * SA_STRIDE + kh * 8);
    v8h ahi = *(const v8h*)(sAb + m * SA_STRIDE + 16 + kh * 8);
    v16h a = __builtin_shufflevector(alo, ahi, 0,1,2,3,4,5,6,7,8,9,10,11,12,13,14,15);

    // preload all 4 B fragments, then issue the 4 WMMAs back-to-back
    v16h bf[4];
#pragma unroll
    for (int nt = 0; nt < 4; ++nt) {
      const _Float16* bp = sBb + (nt * 16 + l16) * SA_STRIDE + kh * 8;
      v8h blo = *(const v8h*)bp;
      v8h bhi = *(const v8h*)(bp + 16);
      bf[nt] = __builtin_shufflevector(blo, bhi, 0,1,2,3,4,5,6,7,8,9,10,11,12,13,14,15);
    }
#pragma unroll
    for (int nt = 0; nt < 4; ++nt) {
      acc[nt] = __builtin_amdgcn_wmma_f32_16x16x32_f16(
          /*neg_a=*/false, a, /*neg_b=*/false, bf[nt],
          /*c_mod=*/(short)0, acc[nt], /*reuse_a=*/false, /*reuse_b=*/false);
    }
    __syncthreads();   // protect buffer `buf` before it is refilled next iter
  }

  // C layout: element r -> M = wave*16 + kh*8 + r, N = colBase + nt*16 + l16
#pragma unroll
  for (int nt = 0; nt < 4; ++nt) {
#pragma unroll
    for (int r = 0; r < 8; ++r) {
      long row = rowBase + wave * 16 + kh * 8 + r;
      int  col = colBase + nt * 16 + l16;
      C[row * (long)N + col] = acc[nt][r];
    }
  }
}

// ---------------------------------------------------------------- SpMM
// agg[dst[e], :] += h[src[e], :]   one thread per (edge, 4-feature chunk)
__global__ void spmm_kernel(const float* __restrict__ h, const int* __restrict__ src,
                            const int* __restrict__ dst, float* __restrict__ agg, int fo) {
  long t = (long)blockIdx.x * blockDim.x + threadIdx.x;
  int chunks = fo >> 2;
  long total = (long)N_EDGES * chunks;
  if (t >= total) return;
  int e = (int)(t / chunks);
  int c = (int)(t % chunks) * 4;
  long s = src[e], d = dst[e];
  float4 v = *(const float4*)(h + s * (long)fo + c);
  float* o = agg + d * (long)fo + c;
  atomicAdd(o + 0, v.x);
  atomicAdd(o + 1, v.y);
  atomicAdd(o + 2, v.z);
  atomicAdd(o + 3, v.w);
}

// ---------------------------------------------------------------- finalize
// layers 1..4: Anext = leaky(agg*dinvI + b) * dinvO  (f16, next layer's K = fo)
__global__ void finalize_mid_kernel(const float* __restrict__ agg, const float* __restrict__ b,
                                    const float* __restrict__ dinvI, const float* __restrict__ dinvO,
                                    _Float16* __restrict__ Anext, int fo) {
  long i = (long)blockIdx.x * blockDim.x + threadIdx.x;
  long total = (long)N_PAD * fo;
  if (i >= total) return;
  int row = (int)(i / fo);
  int k   = (int)(i % fo);
  float v = 0.0f;
  if (row < N_NODES) {
    v = agg[(long)row * fo + k] * dinvI[row] + b[k];
    v = (v > 0.0f) ? v : 0.01f * v;
    v *= dinvO[row];
  }
  Anext[i] = (_Float16)v;
}

// layer 5: out = agg*dinvI + b   (in place on d_out, f32)
__global__ void finalize_last_kernel(float* __restrict__ agg, const float* __restrict__ b,
                                     const float* __restrict__ dinvI, int fo) {
  long i = (long)blockIdx.x * blockDim.x + threadIdx.x;
  long total = (long)N_NODES * fo;
  if (i >= total) return;
  int row = (int)(i / fo);
  int k   = (int)(i % fo);
  agg[i] = agg[i] * dinvI[row] + b[k];
}

// ---------------------------------------------------------------- launcher
static inline long cdiv(long a, long b) { return (a + b - 1) / b; }

extern "C" void kernel_launch(void* const* d_in, const int* in_sizes, int n_in,
                              void* d_out, int out_size, void* d_ws, size_t ws_size,
                              hipStream_t stream) {
  (void)in_sizes; (void)n_in; (void)out_size; (void)ws_size;

  const float* x   = (const float*)d_in[0];
  const int*   ei  = (const int*)d_in[1];     // [2, N_EDGES] int32
  const int*   src = ei;
  const int*   dst = ei + N_EDGES;
  const float* W[5] = {(const float*)d_in[2], (const float*)d_in[4], (const float*)d_in[6],
                       (const float*)d_in[8], (const float*)d_in[10]};
  const float* bb[5] = {(const float*)d_in[3], (const float*)d_in[5], (const float*)d_in[7],
                        (const float*)d_in[9], (const float*)d_in[11]};
  float* out = (float*)d_out;   // also used as the scatter (agg) buffer every layer

  // workspace layout (bytes)
  char* ws = (char*)d_ws;
  float*    dinvO = (float*)(ws + 0);                 // 50000 f32   (pad to 200704)
  float*    dinvI = (float*)(ws + 200704);            // 50000 f32
  _Float16* bufWt = (_Float16*)(ws + 401408);         // <= 524288 halfs (1 MB)
  _Float16* bufA  = (_Float16*)(ws + 1449984);        // N_PAD x 1024 halfs (102.5 MB)
  float*    bufH  = (float*)(ws + 103948288);         // N_PAD x 2048 f32  (410 MB)

  const int fis[5]   = {300, 1024, 512, 256, 128};
  const int fipad[5] = {320, 1024, 512, 256, 128};
  const int fos[5]   = {1024, 512, 256, 128, 2048};

  // --- degrees -> dinv ---
  fill_f32<<<(int)cdiv(N_NODES, 256), 256, 0, stream>>>(dinvO, 0.0f, N_NODES);
  fill_f32<<<(int)cdiv(N_NODES, 256), 256, 0, stream>>>(dinvI, 0.0f, N_NODES);
  degree_kernel<<<(int)cdiv(N_EDGES, 256), 256, 0, stream>>>(src, dst, dinvO, dinvI);
  rsqrt_kernel<<<(int)cdiv(N_NODES, 256), 256, 0, stream>>>(dinvO, N_NODES);
  rsqrt_kernel<<<(int)cdiv(N_NODES, 256), 256, 0, stream>>>(dinvI, N_NODES);

  // --- stage layer-1 input (f16, K-padded, dinv_out pre-applied) ---
  {
    long n = (long)N_PAD * fipad[0];
    stage0_kernel<<<(int)cdiv(n, 256), 256, 0, stream>>>(x, dinvO, bufA, fis[0], fipad[0]);
  }

  for (int L = 0; L < 5; ++L) {
    const int K  = fipad[L];
    const int N  = fos[L];

    // weight -> transposed f16
    {
      long n = (long)N * K;
      wt_kernel<<<(int)cdiv(n, 256), 256, 0, stream>>>(W[L], bufWt, fis[L], N, K);
    }
    // GEMM: bufH = bufA @ W
    {
      dim3 grid(N_PAD / 128, N / 64);
      gemm_wmma_kernel<<<grid, 256, 0, stream>>>(bufA, bufWt, bufH, K, N);
    }
    // zero scatter target, then SpMM
    {
      long n = (long)N_NODES * N;
      fill_f32<<<(int)cdiv(n, 256), 256, 0, stream>>>(out, 0.0f, n);
      long t = (long)N_EDGES * (N >> 2);
      spmm_kernel<<<(int)cdiv(t, 256), 256, 0, stream>>>(bufH, src, dst, out, N);
    }
    // finalize
    if (L < 4) {
      long n = (long)N_PAD * N;
      finalize_mid_kernel<<<(int)cdiv(n, 256), 256, 0, stream>>>(out, bb[L], dinvI, dinvO, bufA, N);
    } else {
      long n = (long)N_NODES * N;
      finalize_last_kernel<<<(int)cdiv(n, 256), 256, 0, stream>>>(out, bb[L], dinvI, N);
    }
  }
}
